// MixtralSparseMoeBlock_43748536877283
// MI455X (gfx1250) — matrix operand order repaired and checked by reading, not compiled, not run on hardware
//
#include <hip/hip_runtime.h>
#include <hip/hip_bf16.h>

// ---------------------------------------------------------------------------
// Mixtral sparse MoE block for gfx1250 (MI455X), wave32 + WMMA bf16.
//   x[T,D] f32, gate_w[E,D], W1/W3[E,D,F], W2[E,F,D]  (all f32)
//   out = concat( out[T,D] f32 , router_logits[T,E] f32 )
// T=8192 D=1024 F=4096 E=8 TOPK=2
//
// TM=32 tokens per block: each B tile loaded into registers feeds TWO wmma
// (one per 16-row A subtile) -> 32 FLOP/byte vs L2 on the weight stream.
// ---------------------------------------------------------------------------

typedef __bf16 bf16;
typedef __attribute__((ext_vector_type(16))) __bf16 v16bf;
typedef __attribute__((ext_vector_type(8)))  __bf16 v8bf;
typedef __attribute__((ext_vector_type(8)))  float  v8f;

#define T_TOK 8192
#define DDIM  1024
#define FDIM  4096
#define NEXP  8
#define TM    32      // tokens per tile (2 WMMA row-subtiles)
#define FCH   128     // F-chunk per outer iteration (8 waves * 16 cols)
#define TILES_PER_E (T_TOK / TM)   // 256

__device__ __forceinline__ v16bf cat8(v8bf lo, v8bf hi) {
    return __builtin_shufflevector(lo, hi, 0,1,2,3,4,5,6,7,8,9,10,11,12,13,14,15);
}

// ---------------------------------------------------------------------------
// 0) zero the dense output region + per-expert counters
// ---------------------------------------------------------------------------
__global__ __launch_bounds__(256)
void init_kernel(float* __restrict__ out, int* __restrict__ counts) {
    size_t i = (size_t)blockIdx.x * 256 + threadIdx.x;
    if (i < (size_t)T_TOK * DDIM) out[i] = 0.0f;
    if (i < NEXP) counts[i] = 0;
}

// ---------------------------------------------------------------------------
// 1) f32 [K][N] -> bf16 [N][K] transpose-convert (per expert in blockIdx.z)
// ---------------------------------------------------------------------------
__global__ __launch_bounds__(256)
void transpose_cvt_kernel(const float* __restrict__ in, bf16* __restrict__ out,
                          int K, int N) {
    __shared__ float tile[32][33];
    const float* src = in  + (size_t)blockIdx.z * K * N;
    bf16*        dst = out + (size_t)blockIdx.z * K * N;
    int n0 = blockIdx.x * 32, k0 = blockIdx.y * 32;
    int c = threadIdx.x & 31, r8 = threadIdx.x >> 5;
#pragma unroll
    for (int i = 0; i < 4; ++i) {
        int r = r8 + i * 8;
        tile[r][c] = src[(size_t)(k0 + r) * N + n0 + c];
    }
    __syncthreads();
#pragma unroll
    for (int i = 0; i < 4; ++i) {
        int r = r8 + i * 8;                       // output row within N-tile
        dst[(size_t)(n0 + r) * K + k0 + c] = (bf16)tile[c][r];
    }
}

// ---------------------------------------------------------------------------
// 2) router: logits, softmax, top-2, compaction into per-expert lists
//    one wave (32 lanes) per token
// ---------------------------------------------------------------------------
__global__ __launch_bounds__(256)
void router_kernel(const float* __restrict__ x, const float* __restrict__ gate_w,
                   float* __restrict__ logits_out,
                   int* __restrict__ counts, int* __restrict__ tok_idx,
                   float* __restrict__ tok_w) {
    __shared__ float gws[NEXP][DDIM];             // 32 KB
    int tid = threadIdx.x;
    for (int i = tid; i < NEXP * DDIM / 4; i += 256)
        ((float4*)&gws[0][0])[i] = ((const float4*)gate_w)[i];
    __syncthreads();

    int lane = tid & 31, wave = tid >> 5;
    int t = blockIdx.x * 8 + wave;
    const float* xr = x + (size_t)t * DDIM;

    float acc[NEXP] = {};
#pragma unroll
    for (int i = 0; i < DDIM / (32 * 4); ++i) {   // 8 iterations
        int d = (i * 32 + lane) * 4;
        float4 xv = *(const float4*)&xr[d];
#pragma unroll
        for (int e = 0; e < NEXP; ++e) {
            float4 gv = *(const float4*)&gws[e][d];
            acc[e] += xv.x * gv.x + xv.y * gv.y + xv.z * gv.z + xv.w * gv.w;
        }
    }
#pragma unroll
    for (int off = 16; off > 0; off >>= 1)
#pragma unroll
        for (int e = 0; e < NEXP; ++e) acc[e] += __shfl_xor(acc[e], off, 32);

    if (lane == 0) {
        float m = acc[0];
#pragma unroll
        for (int e = 1; e < NEXP; ++e) m = fmaxf(m, acc[e]);
        float p[NEXP];
#pragma unroll
        for (int e = 0; e < NEXP; ++e) p[e] = __expf(acc[e] - m);
        // top-2, first-occurrence tie-break (matches jax top_k)
        int i0 = 0;
#pragma unroll
        for (int e = 1; e < NEXP; ++e) if (p[e] > p[i0]) i0 = e;
        int i1 = (i0 == 0) ? 1 : 0;
#pragma unroll
        for (int e = 0; e < NEXP; ++e) if (e != i0 && p[e] > p[i1]) i1 = e;
        float s = p[i0] + p[i1];
        float w0 = p[i0] / s, w1 = p[i1] / s;
#pragma unroll
        for (int e = 0; e < NEXP; ++e) logits_out[(size_t)t * NEXP + e] = acc[e];
        int pos0 = atomicAdd(&counts[i0], 1);
        tok_idx[i0 * T_TOK + pos0] = t;  tok_w[i0 * T_TOK + pos0] = w0;
        int pos1 = atomicAdd(&counts[i1], 1);
        tok_idx[i1 * T_TOK + pos1] = t;  tok_w[i1 * T_TOK + pos1] = w1;
    }
}

// ---------------------------------------------------------------------------
// 3) fused expert FFN: Y = (silu(X W1) * (X W3)) W2, gathered 32-token tiles
//    256 threads = 8 waves; WMMA f32_16x16x32_bf16 throughout.
//    Weights are pre-transposed bf16 [N][K]: B operand per lane = 16
//    K elements as two runs of 8 at 8h and 16+8h (two b128 loads), reused for both M-subtiles.
// ---------------------------------------------------------------------------
__global__ __launch_bounds__(256)
void moe_expert_kernel(const float* __restrict__ x,
                       const bf16* __restrict__ W1t,   // [E][F][D]
                       const bf16* __restrict__ W3t,   // [E][F][D]
                       const bf16* __restrict__ W2t,   // [E][D][F]
                       const int*  __restrict__ counts,
                       const int*  __restrict__ tok_idx,
                       const float* __restrict__ tok_w,
                       float* __restrict__ out) {
    int e    = blockIdx.x >> 8;               // / TILES_PER_E (256)
    int tile = blockIdx.x & (TILES_PER_E - 1);
    int cnt  = counts[e];
    if (tile * TM >= cnt) return;             // block-uniform: EXEC stays full

    // dynamic LDS: Xs[32][1024] bf16 (64KB) | Hs[32][128] bf16 (8KB) | tails
    extern __shared__ char smem[];
    bf16  (*Xs)[DDIM] = (bf16 (*)[DDIM])smem;
    bf16  (*Hs)[FCH]  = (bf16 (*)[FCH])(smem + TM * DDIM * sizeof(bf16));
    int*   rowTok = (int*)(smem + TM * DDIM * sizeof(bf16) + TM * FCH * sizeof(bf16));
    float* rowW   = (float*)(rowTok + TM);

    int tid = threadIdx.x;
    if (tid < TM) {
        int slot  = tile * TM + tid;
        int cslot = (slot < cnt) ? slot : (cnt - 1);
        rowTok[tid] = tok_idx[e * T_TOK + cslot];
        rowW[tid]   = (slot < cnt) ? tok_w[e * T_TOK + cslot] : 0.0f;
    }
    __syncthreads();

    // cooperative gather: 32 rows x 1024 f32 -> bf16 LDS
    for (int i = tid; i < TM * DDIM / 4; i += 256) {
        int r  = i / (DDIM / 4);
        int c4 = i % (DDIM / 4);
        float4 v = ((const float4*)(x + (size_t)rowTok[r] * DDIM))[c4];
        bf16* dst = &Xs[r][c4 * 4];
        dst[0] = (bf16)v.x; dst[1] = (bf16)v.y;
        dst[2] = (bf16)v.z; dst[3] = (bf16)v.w;
    }
    __syncthreads();

    int lane = tid & 31, wave = tid >> 5;
    int half = lane >> 4, n16 = lane & 15;

    const bf16* w1e = W1t + (size_t)e * FDIM * DDIM;
    const bf16* w3e = W3t + (size_t)e * FDIM * DDIM;
    const bf16* w2e = W2t + (size_t)e * DDIM * FDIM;

    v8f yacc[2][8] = {};                      // 32 x 128 slice of Y (128 VGPRs)

    for (int fo = 0; fo < FDIM / FCH; ++fo) {        // 32 outer iterations
        // ---- GEMM1 & GEMM3: this wave's 32x16 H tile, K = D = 1024 ----
        int ncol = fo * FCH + wave * 16 + n16;       // W1/W3 column (row of Wt)
        const bf16* b1 = w1e + (size_t)ncol * DDIM;
        const bf16* b3 = w3e + (size_t)ncol * DDIM;
        v8f acc1[2] = {}, acc3[2] = {};
        for (int kk = 0; kk < DDIM / 32; ++kk) {
            int kb = kk * 32;
            // A subtiles, ISA 16-bit A layout: lanes<16 hold K {0..7,16..23}
            v16bf a0 = cat8(*(const v8bf*)&Xs[n16][kb + half * 8],
                            *(const v8bf*)&Xs[n16][kb + 16 + half * 8]);
            v16bf a1 = cat8(*(const v8bf*)&Xs[16 + n16][kb + half * 8],
                            *(const v8bf*)&Xs[16 + n16][kb + 16 + half * 8]);
            // B tiles: K {0..7,16..23} (lanes<16) at this N column; reused by a0 and a1
            v16bf bb1 = cat8(*(const v8bf*)&b1[kb + half * 8],
                             *(const v8bf*)&b1[kb + 16 + half * 8]);
            acc1[0] = __builtin_amdgcn_wmma_f32_16x16x32_bf16(
                          false, a0, false, bb1, (short)0, acc1[0], false, false);
            acc1[1] = __builtin_amdgcn_wmma_f32_16x16x32_bf16(
                          false, a1, false, bb1, (short)0, acc1[1], false, false);
            v16bf bb3 = cat8(*(const v8bf*)&b3[kb + half * 8],
                             *(const v8bf*)&b3[kb + 16 + half * 8]);
            acc3[0] = __builtin_amdgcn_wmma_f32_16x16x32_bf16(
                          false, a0, false, bb3, (short)0, acc3[0], false, false);
            acc3[1] = __builtin_amdgcn_wmma_f32_16x16x32_bf16(
                          false, a1, false, bb3, (short)0, acc3[1], false, false);
        }

        __syncthreads();               // previous Hs chunk fully consumed
        // silu(acc1) * acc3 -> bf16 Hs ; C layout: n=lane&15, m=r+half*8
#pragma unroll
        for (int sub = 0; sub < 2; ++sub)
#pragma unroll
            for (int r = 0; r < 8; ++r) {
                float s = acc1[sub][r];
                float h = (s / (1.0f + __expf(-s))) * acc3[sub][r];
                Hs[sub * 16 + r + half * 8][wave * 16 + n16] = (bf16)h;
            }
        __syncthreads();

        // ---- GEMM2 partial: Y += H_chunk * W2[kglob:kglob+128, :] ----
        for (int kk = 0; kk < FCH / 32; ++kk) {
            int kb = kk * 32;
            v16bf a0 = cat8(*(const v8bf*)&Hs[n16][kb + half * 8],
                            *(const v8bf*)&Hs[n16][kb + 16 + half * 8]);
            v16bf a1 = cat8(*(const v8bf*)&Hs[16 + n16][kb + half * 8],
                            *(const v8bf*)&Hs[16 + n16][kb + 16 + half * 8]);
            int kglob = fo * FCH + kb;
#pragma unroll
            for (int j = 0; j < 8; ++j) {
                int nc = wave * 128 + j * 16 + n16;   // output col (row of W2t)
                const bf16* b2 = w2e + (size_t)nc * FDIM + kglob;
                v16bf bb = cat8(*(const v8bf*)&b2[half * 8],
                                *(const v8bf*)&b2[16 + half * 8]);
                yacc[0][j] = __builtin_amdgcn_wmma_f32_16x16x32_bf16(
                                 false, a0, false, bb, (short)0, yacc[0][j], false, false);
                yacc[1][j] = __builtin_amdgcn_wmma_f32_16x16x32_bf16(
                                 false, a1, false, bb, (short)0, yacc[1][j], false, false);
            }
        }
    }

    // scatter: each output element gets exactly 2 commutative float adds
#pragma unroll
    for (int sub = 0; sub < 2; ++sub)
#pragma unroll
        for (int j = 0; j < 8; ++j) {
            int nc = wave * 128 + j * 16 + n16;
#pragma unroll
            for (int r = 0; r < 8; ++r) {
                int m = sub * 16 + r + half * 8;
                float v = yacc[sub][j][r] * rowW[m];
                atomicAdd(&out[(size_t)rowTok[m] * DDIM + nc], v);
            }
        }
}

// ---------------------------------------------------------------------------
// host-side launcher
// ---------------------------------------------------------------------------
extern "C" void kernel_launch(void* const* d_in, const int* in_sizes, int n_in,
                              void* d_out, int out_size, void* d_ws, size_t ws_size,
                              hipStream_t stream) {
    const float* x      = (const float*)d_in[0];
    const float* gate_w = (const float*)d_in[1];
    const float* W1     = (const float*)d_in[2];
    const float* W3     = (const float*)d_in[3];
    const float* W2     = (const float*)d_in[4];

    float* out    = (float*)d_out;                       // [T*D]
    float* logits = out + (size_t)T_TOK * DDIM;          // [T*E]

    // workspace layout
    char*  ws      = (char*)d_ws;
    int*   counts  = (int*)ws;                                           // 32 B
    int*   tok_idx = (int*)(ws + 256);                                   // 256 KB
    float* tok_w   = (float*)(ws + 256 + (size_t)NEXP * T_TOK * 4);      // 256 KB
    size_t woff    = 1u << 20;                                           // 1 MB
    size_t wmat    = (size_t)NEXP * FDIM * DDIM;                         // elems
    bf16*  W1t = (bf16*)(ws + woff);
    bf16*  W3t = (bf16*)(ws + woff + wmat * sizeof(bf16));
    bf16*  W2t = (bf16*)(ws + woff + 2 * wmat * sizeof(bf16));

    // dynamic LDS for the fused FFN kernel (exceeds 64KB default opt-in)
    const int smemBytes = TM * DDIM * (int)sizeof(bf16)     // Xs 64KB
                        + TM * FCH * (int)sizeof(bf16)      // Hs  8KB
                        + TM * (int)sizeof(int)             // rowTok
                        + TM * (int)sizeof(float);          // rowW
    (void)hipFuncSetAttribute((const void*)moe_expert_kernel,
                              hipFuncAttributeMaxDynamicSharedMemorySize,
                              smemBytes);

    // 0) zero dense output + counters
    init_kernel<<<(T_TOK * DDIM + 255) / 256, 256, 0, stream>>>(out, counts);

    // 1) weight transpose-convert to bf16 [N][K]
    transpose_cvt_kernel<<<dim3(FDIM / 32, DDIM / 32, NEXP), 256, 0, stream>>>(
        W1, W1t, DDIM, FDIM);
    transpose_cvt_kernel<<<dim3(FDIM / 32, DDIM / 32, NEXP), 256, 0, stream>>>(
        W3, W3t, DDIM, FDIM);
    transpose_cvt_kernel<<<dim3(DDIM / 32, FDIM / 32, NEXP), 256, 0, stream>>>(
        W2, W2t, FDIM, DDIM);

    // 2) routing + token compaction
    router_kernel<<<T_TOK / 8, 256, 0, stream>>>(x, gate_w, logits,
                                                 counts, tok_idx, tok_w);

    // 3) fused expert FFN (expert-major grid -> per-expert weight reuse in L2)
    moe_expert_kernel<<<NEXP * TILES_PER_E, 256, smemBytes, stream>>>(
        x, W1t, W3t, W2t, counts, tok_idx, tok_w, out);
}
